// AutoElmanCell_29497835388992
// MI455X (gfx1250) — compile-verified, matrix operand order checked by reading, not run on hardware
//
#include <hip/hip_runtime.h>
#include <hip/hip_bf16.h>
#include <stdint.h>

// ---------------------------------------------------------------------------
// Problem constants (reference: T=2048, B=8, D=1024)
// ---------------------------------------------------------------------------
#define TT 2048
#define BB 8
#define DD 1024
#define M_GATE (TT * BB)          // 16384
#define HSTRIDE (BB * DD)         // 8192 floats per h timestep

typedef float    v8f  __attribute__((ext_vector_type(8)));
typedef float    v4f  __attribute__((ext_vector_type(4)));
typedef float    v2f  __attribute__((ext_vector_type(2)));
typedef __bf16   v16bf __attribute__((ext_vector_type(16)));
typedef __bf16   v8bf  __attribute__((ext_vector_type(8)));
typedef __bf16   v4bf  __attribute__((ext_vector_type(4)));

// fp32 -> bf16, round-to-nearest-even (bit manipulation: no reliance on
// __bf16 conversion support in the target libm)
__device__ __forceinline__ __bf16 f2bf(float f) {
    unsigned u = __builtin_bit_cast(unsigned, f);
    unsigned r = u + 0x7FFFu + ((u >> 16) & 1u);
    unsigned short h = (unsigned short)(r >> 16);
    return __builtin_bit_cast(__bf16, h);
}

// ---------------------------------------------------------------------------
// Kernel 0: init — zero per-step barrier counters, copy h0 -> h[0]
// ---------------------------------------------------------------------------
__global__ __launch_bounds__(256) void init_kernel(const float* __restrict__ h0,
                                                   float* __restrict__ h,
                                                   unsigned* __restrict__ cnt) {
    int i = blockIdx.x * 256 + threadIdx.x;   // grid: 32x256 = 8192
    if (i < TT) cnt[i] = 0u;
    if (i < HSTRIDE) h[i] = h0[i];
}

// ---------------------------------------------------------------------------
// Kernel 1: gate GEMM  (M=16384, N=1024, K=1024), bf16 WMMA, fp32 accum.
// Block tile 128x128, BK=64 (two 16x16x32 k-substeps per stage), 8 waves
// (4 along M x 2 along N), per-wave 2x4 = 8 accum tiles. Double-buffered LDS
// with fp32->bf16 conversion in flight, plus global_prefetch two stages ahead.
// Writes silu(x@Wg.T + bg) into `gate_out` (= output region of d_out).
// ---------------------------------------------------------------------------
#define GLD 72   // LDS row stride in bf16 elements (64 + 8 pad) -> 144B rows

__device__ __forceinline__ v16bf load_frag(const __bf16* tile, int rowBase, int lane) {
    // tile row-major, stride GLD; fragment per ISA 16-bit A/B layout:
    //   lanes 0-15: row=lane,    k = 0..7  and 16..23
    //   lanes16-31: row=lane-16, k = 8..15 and 24..31
    int r  = rowBase + (lane & 15);
    int kb = (lane >> 4) << 3;
    const __bf16* p = tile + r * GLD + kb;
    v8bf lo = *(const v8bf*)(p);
    v8bf hi = *(const v8bf*)(p + 16);
    return __builtin_shufflevector(lo, hi, 0,1,2,3,4,5,6,7,8,9,10,11,12,13,14,15);
}

__global__ __launch_bounds__(256) void gate_gemm_kernel(const float* __restrict__ x,
                                                        const float* __restrict__ Wg,
                                                        const float* __restrict__ bg,
                                                        float* __restrict__ gate_out) {
    __shared__ __align__(16) __bf16 As[2][128 * GLD];
    __shared__ __align__(16) __bf16 Bs[2][128 * GLD];

    const int t    = threadIdx.x;
    const int lane = t & 31;
    const int wid  = t >> 5;
    const int wm   = wid & 3;        // wave M index: 0..3 (32 rows each)
    const int wn   = wid >> 2;       // wave N index: 0..1 (64 cols each)
    const int row0 = blockIdx.y * 128;
    const int col0 = blockIdx.x * 128;

    v8f acc[2][4];
#pragma unroll
    for (int i = 0; i < 2; ++i)
#pragma unroll
        for (int j = 0; j < 4; ++j) acc[i][j] = (v8f){0,0,0,0,0,0,0,0};

    // per-lane bias (column depends only on lane&15 and the n-tile)
    float bgv[4];
#pragma unroll
    for (int j = 0; j < 4; ++j)
        bgv[j] = bg[col0 + wn * 64 + j * 16 + (lane & 15)];

    auto stage = [&](int buf, int k0) {
        // 128 rows x 64 cols of fp32 per tile; 4 threads/row, 4 x float4 each,
        // two row-passes. Convert to bf16 into LDS. Prefetch two stages ahead
        // (double buffering already hides one stage) -> global_prefetch_b8.
        const int r = t >> 2;
        const int q = t & 3;
#pragma unroll
        for (int p = 0; p < 2; ++p) {
            const int rr = r + p * 64;
            const float* ax = x  + (size_t)(row0 + rr) * DD + k0 + q * 16;
            const float* bx = Wg + (size_t)(col0 + rr) * DD + k0 + q * 16;
            if (k0 + 128 < DD) {
                __builtin_prefetch(ax + 128, 0, 0);
                __builtin_prefetch(bx + 128, 0, 0);
            }
#pragma unroll
            for (int j = 0; j < 4; ++j) {
                v4f av = *(const v4f*)(ax + j * 4);
                v4f bv = *(const v4f*)(bx + j * 4);
                v4bf ac, bc;
#pragma unroll
                for (int e = 0; e < 4; ++e) { ac[e] = f2bf(av[e]); bc[e] = f2bf(bv[e]); }
                *(v4bf*)&As[buf][rr * GLD + q * 16 + j * 4] = ac;
                *(v4bf*)&Bs[buf][rr * GLD + q * 16 + j * 4] = bc;
            }
        }
    };

    auto compute = [&](int buf) {
#pragma unroll
        for (int ks = 0; ks < 2; ++ks) {          // two 16x16x32 substeps in BK=64
            v16bf af[2], bfv[4];
#pragma unroll
            for (int i = 0; i < 2; ++i)
                af[i] = load_frag(&As[buf][ks * 32], wm * 32 + i * 16, lane);
#pragma unroll
            for (int j = 0; j < 4; ++j)
                bfv[j] = load_frag(&Bs[buf][ks * 32], wn * 64 + j * 16, lane);
#pragma unroll
            for (int i = 0; i < 2; ++i)
#pragma unroll
                for (int j = 0; j < 4; ++j)
                    acc[i][j] = __builtin_amdgcn_wmma_f32_16x16x32_bf16(
                        false, af[i], false, bfv[j], (short)0, acc[i][j], false, false);
        }
    };

    stage(0, 0);
    const int NK = DD / 64;                       // 16 stages
#pragma unroll 1
    for (int kt = 0; kt < NK; ++kt) {
        __syncthreads();
        if (kt + 1 < NK) stage((kt + 1) & 1, (kt + 1) * 64);
        compute(kt & 1);
    }

    // Epilogue: bias + silu, scatter-store fp32.
#pragma unroll
    for (int i = 0; i < 2; ++i) {
#pragma unroll
        for (int j = 0; j < 4; ++j) {
            const int cg = col0 + wn * 64 + j * 16 + (lane & 15);
#pragma unroll
            for (int e = 0; e < 8; ++e) {
                const int rg = row0 + wm * 32 + i * 16 + e + 8 * (lane >> 4);
                float z = acc[i][j][e] + bgv[j];
                float s = z / (1.0f + __expf(-z));          // silu
                gate_out[(size_t)rg * DD + cg] = s;
            }
        }
    }
}

// ---------------------------------------------------------------------------
// Kernel 2: autonomous recurrence, persistent across all 2048 steps.
// 64 WGs x 256 threads. WG b owns output columns [16b, 16b+16). Its W_h slice
// (16 x 1024 fp32 = 64 KB) lives in LDS for the whole kernel, filled ONCE via
// CDNA5 async copy (global_load_async_to_lds_b128 + s_wait_asynccnt): no VGPR
// staging. LDS layout [k/4][n][4 floats]: each lane's 16B is contiguous
// (async requirement), global fill is fully coalesced (lane = consecutive
// k-quad of one W_h row), and B-fragment ds_load_b64 stays conflict-free
// (lane n -> banks 4n / 4n+2).
// Each of the 8 waves covers K-chunk of 128 -> 32 chained V_WMMA_F32_16X16X4_F32
// (full fp32: error compounds through 2048 tanh steps; this path is
// latency-bound so fp32 WMMA costs nothing vs bf16).
// Per step: partial-tile reduce via LDS, tanh+bias, store h slice, then a
// device-scope counter barrier (cnt[step] in ws, zeroed by init each launch).
// ---------------------------------------------------------------------------
#define NWG_REC 64

__global__ __launch_bounds__(256) void recurrence_kernel(const float* __restrict__ Wh,
                                                         const float* __restrict__ bh,
                                                         float* __restrict__ h,
                                                         unsigned* __restrict__ cnt) {
    __shared__ __align__(16) float Whs[256 * 16 * 4];   // [kq][n][4] = 64 KB
    __shared__ float part[8 * 256];                     // per-wave partial tiles

    const int t    = threadIdx.x;
    const int lane = t & 31;
    const int wid  = t >> 5;
    const int n0   = blockIdx.x * 16;

    // ---- async fill of the W_h slice ------------------------------------
    // iteration `it` = row n0+it; lane t = k-quad t (16B). Flat->LDS address:
    // low 32 bits of a __shared__ pointer are the wave-relative LDS offset
    // (ISA 10.2).
    {
        const unsigned ldsbase = (unsigned)(uintptr_t)(void*)&Whs[0];
#pragma unroll 1
        for (int it = 0; it < 16; ++it) {
            const float* src = Wh + (size_t)(n0 + it) * DD + t * 4;
            unsigned dst = ldsbase + (unsigned)(t * 16 + it) * 16u;
            asm volatile("global_load_async_to_lds_b128 %0, %1, off"
                         :: "v"(dst), "v"(src) : "memory");
        }
        asm volatile("s_wait_asynccnt 0x0" ::: "memory");
    }
    __syncthreads();

    const int  m      = lane & 15;          // A row (batch), rows 8..15 are pad
    const int  n      = lane & 15;          // B column within slice
    const int  koff   = (lane >> 4) * 2;    // K sub-offset per ISA f32 layout
    const bool mvalid = (m < BB);
    const float bhv   = bh[n0 + (t & 15)];  // bias for reduce phase (col = t&15)
    const int  kbase  = wid * 128;          // this wave's K-chunk

#pragma unroll 1
    for (int step = 0; step < TT; ++step) {
        const float* hp = h + (size_t)step * HSTRIDE;
        v8f acc = (v8f){0,0,0,0,0,0,0,0};
#pragma unroll
        for (int q = 0; q < 32; ++q) {
            const int k = kbase + q * 4;
            v2f a = (v2f){0.f, 0.f};
            if (mvalid) a = *(const v2f*)(hp + m * DD + k + koff);
            v2f b = *(const v2f*)&Whs[((k >> 2) * 16 + n) * 4 + koff];
            acc = __builtin_amdgcn_wmma_f32_16x16x4_f32(
                false, a, false, b, (short)0, acc, false, false);
        }
        // write per-wave partial tile (C layout: VGPR e -> row e+8*(lane/16))
#pragma unroll
        for (int e = 0; e < 8; ++e)
            part[wid * 256 + (e + 8 * (lane >> 4)) * 16 + (lane & 15)] = acc[e];
        __syncthreads();

        // reduce 8 partials, bias, tanh, store h[step+1] slice
        {
            float s = 0.f;
#pragma unroll
            for (int w = 0; w < 8; ++w) s += part[w * 256 + t];
            float val = tanhf(s + bhv);
            int rr = t >> 4, cc = t & 15;
            if (rr < BB)
                h[(size_t)(step + 1) * HSTRIDE + rr * DD + n0 + cc] = val;
        }
        __threadfence();      // make this thread's h stores device-visible
        __syncthreads();      // all waves' stores fenced before arrive

        if (t == 0) {
            __hip_atomic_fetch_add(&cnt[step], 1u, __ATOMIC_RELEASE,
                                   __HIP_MEMORY_SCOPE_AGENT);
            while (__hip_atomic_load(&cnt[step], __ATOMIC_ACQUIRE,
                                     __HIP_MEMORY_SCOPE_AGENT) < (unsigned)NWG_REC) {
                __builtin_amdgcn_s_sleep(1);
            }
        }
        __syncthreads();      // barrier release: everyone sees fresh h
    }
}

// ---------------------------------------------------------------------------
// Kernel 3: output[t] = gate[t] * h[t+1], in place over the gate buffer.
// Pure bandwidth (3 x 67 MB -> ~9 us at 23.3 TB/s).
// ---------------------------------------------------------------------------
__global__ __launch_bounds__(256) void mul_kernel(float* __restrict__ out,
                                                  const float* __restrict__ h) {
    size_t i = ((size_t)blockIdx.x * 256 + threadIdx.x);   // float4 index
    v4f g  = *(const v4f*)(out + i * 4);
    v4f hh = *(const v4f*)(h + HSTRIDE + i * 4);           // h shifted one step
    v4f r  = g * hh;
    *(v4f*)(out + i * 4) = r;
}

// ---------------------------------------------------------------------------
// Launch
// ---------------------------------------------------------------------------
extern "C" void kernel_launch(void* const* d_in, const int* in_sizes, int n_in,
                              void* d_out, int out_size, void* d_ws, size_t ws_size,
                              hipStream_t stream) {
    const float* x      = (const float*)d_in[0];
    const float* h0     = (const float*)d_in[1];
    const float* W_h    = (const float*)d_in[2];
    const float* W_gate = (const float*)d_in[3];
    const float* b_h    = (const float*)d_in[4];
    const float* b_gate = (const float*)d_in[5];

    float* out = (float*)d_out;                       // [T,B,D] output
    float* h   = out + (size_t)M_GATE * DD;           // [T+1,B,D] hidden states
    unsigned* cnt = (unsigned*)d_ws;                  // TT barrier counters

    init_kernel<<<dim3(32), dim3(256), 0, stream>>>(h0, h, cnt);
    gate_gemm_kernel<<<dim3(DD / 128, M_GATE / 128), dim3(256), 0, stream>>>(
        x, W_gate, b_gate, out);
    recurrence_kernel<<<dim3(NWG_REC), dim3(256), 0, stream>>>(W_h, b_h, h, cnt);
    mul_kernel<<<dim3((M_GATE * DD) / (256 * 4)), dim3(256), 0, stream>>>(out, h);
}
